// Attention_61701500174555
// MI455X (gfx1250) — matrix operand order repaired and checked by reading, not compiled, not run on hardware
//
#include <hip/hip_runtime.h>
#include <hip/hip_bf16.h>
#include <hip/hip_fp16.h>

// MI455X / gfx1250 (CDNA5, wave32). Attention(Q,K,V) + residual + BN + Swish.
//
// Kernel 1: one workgroup per (b,c) slice (1024 blocks, 4 waves).
//   - Async-copy f32 V slice into LDS (GLOBAL_LOAD_ASYNC_TO_LDS_B128,
//     ASYNCcnt) while staging Q,K (transposed, bf16) + V (bf16) via float4
//     loads + converts.
//   - Per wave: 16-row block of S = Q^T K via v_wmma_f32_16x16x32_bf16
//     (K-dim == H == 32, so one WMMA per 16x16 tile), register softmax,
//     P (bf16) -> LDS, then out^T = P V^T with 16 more WMMAs.
//   - x = out + V staged in LDS; stored with GLOBAL_STORE_ASYNC_FROM_LDS_B128
//     while per-slice sum/sumsq partials are reduced -> ws.
// Kernel 2: per (b,c) slice, fold per-channel stats, normalize + swish.

typedef __attribute__((ext_vector_type(16))) __bf16 v16bf;
typedef __attribute__((ext_vector_type(8)))  __bf16 v8bf;
typedef __attribute__((ext_vector_type(4)))  __bf16 v4bf;
typedef __attribute__((ext_vector_type(8)))  float  v8f;
typedef __attribute__((ext_vector_type(4)))  int    v4i;

#define BATCH  16
#define CH     64
#define HH     32
#define WW     256
#define SLICE  (HH * WW)        // 8192 elements per (b,c)
#define NSLICE (BATCH * CH)     // 1024 slices

#define AS1 __attribute__((address_space(1)))
#define AS3 __attribute__((address_space(3)))

#if defined(__AMDGCN__) && __has_builtin(__builtin_amdgcn_global_load_async_to_lds_b128)
#define HAVE_ASYNC_LOAD 1
#else
#define HAVE_ASYNC_LOAD 0
#endif
#if defined(__AMDGCN__) && __has_builtin(__builtin_amdgcn_global_store_async_from_lds_b128)
#define HAVE_ASYNC_STORE 1
#else
#define HAVE_ASYNC_STORE 0
#endif

// A-operand (16-bit, 16x32): lanes 0-15 hold M=lane, K=0..7 & 16..23 ->
// bytes [0,16) and [32,48) of a 64B row; lanes 16-31 hold K=8..15 & 24..31.
__device__ __forceinline__ v16bf lds_ld_a(const __bf16* row, int half) {
  const char* p = (const char*)row;
  v8bf lo = *(const v8bf*)(p + (half ? 16 : 0));
  v8bf hi = *(const v8bf*)(p + (half ? 48 : 32));
  v16bf r;
#pragma unroll
  for (int i = 0; i < 8; ++i) { r[i] = lo[i]; r[i + 8] = hi[i]; }
  return r;
}

// B-operand (16-bit, 32x16): lanes 0-15 hold N=lane, K=0..7 & 16..23 (bytes [0,16),[32,48));
// lanes 16-31 hold K=8..15 & 24..31 (bytes [16,32),[48,64)) of a 64B "column" row.
__device__ __forceinline__ v16bf lds_ld_b(const __bf16* row, int half) {
  const char* p = (const char*)row + (half ? 16 : 0);
  v8bf lo = *(const v8bf*)(p);
  v8bf hi = *(const v8bf*)(p + 32);
  v16bf r;
#pragma unroll
  for (int i = 0; i < 8; ++i) { r[i] = lo[i]; r[i + 8] = hi[i]; }
  return r;
}

__device__ __forceinline__ v8f wmma_bf16(v16bf a, v16bf b, v8f c) {
  return __builtin_amdgcn_wmma_f32_16x16x32_bf16(
      /*neg_a=*/false, a, /*neg_b=*/false, b,
      /*c_mod=*/(short)0, c, /*reuse_a=*/false, /*reuse_b=*/false);
}

__global__ __launch_bounds__(128) void attn_phase1(
    const float* __restrict__ Q, const float* __restrict__ K,
    const float* __restrict__ V, float* __restrict__ Xout,
    float* __restrict__ ws) {
  __shared__ __align__(16) __bf16 sQt[WW][HH];     // 16 KB, A of gemm1: [w][h]
  __shared__ __align__(16) __bf16 sKt[WW][HH];     // 16 KB, B of gemm1: [v][h]
  __shared__ __align__(16) __bf16 sVb[HH][WW];     // 16 KB, B of gemm2: [h][v]
  __shared__ __align__(16) float  sVf[HH][WW];     // 32 KB, f32 V for residual
  __shared__ __align__(16) __bf16 sP[4][16][WW];   // 32 KB, per-wave P rows
  __shared__ __align__(16) float  sX[HH][WW];      // 32 KB, x staging
  __shared__ float sRed[8];

  const int tid   = threadIdx.x;
  const int wv    = tid >> 5;
  const int lane  = tid & 31;
  const int half  = (lane >> 4) & 1;
  const int l16   = lane & 15;
  const int slice = blockIdx.x;

  const float* q = Q + (size_t)slice * SLICE;
  const float* k = K + (size_t)slice * SLICE;
  const float* v = V + (size_t)slice * SLICE;

#if HAVE_ASYNC_LOAD
  // Kick off the f32 V -> LDS copy asynchronously (pure byte move, ASYNCcnt).
  // It overlaps with the convert/transpose staging below.
  {
    char* vdst = (char*)&sVf[0][0];
    for (int i = tid; i < SLICE / 4; i += 128) {
      __builtin_amdgcn_global_load_async_to_lds_b128(
          (AS1 v4i*)(v + (size_t)i * 4),
          (AS3 v4i*)(vdst + (size_t)i * 16), 0, 0);
    }
  }
#endif

  // Cooperative, coalesced float4 load + bf16 down-convert + transpose.
  {
    const float4* q4 = (const float4*)q;
    const float4* k4 = (const float4*)k;
    const float4* v4 = (const float4*)v;
    for (int i = tid; i < SLICE / 4; i += 128) {
      int h = i >> 6;             // 64 float4 per h-row
      int w = (i & 63) * 4;
      float4 qf = q4[i], kf = k4[i], vf = v4[i];
      sQt[w + 0][h] = (__bf16)qf.x;
      sQt[w + 1][h] = (__bf16)qf.y;
      sQt[w + 2][h] = (__bf16)qf.z;
      sQt[w + 3][h] = (__bf16)qf.w;
      sKt[w + 0][h] = (__bf16)kf.x;
      sKt[w + 1][h] = (__bf16)kf.y;
      sKt[w + 2][h] = (__bf16)kf.z;
      sKt[w + 3][h] = (__bf16)kf.w;
      v4bf vb;
      vb[0] = (__bf16)vf.x; vb[1] = (__bf16)vf.y;
      vb[2] = (__bf16)vf.z; vb[3] = (__bf16)vf.w;
      *(v4bf*)&sVb[h][w] = vb;    // row-contiguous: single ds_store_b64
#if !HAVE_ASYNC_LOAD
      *(float4*)&sVf[h][w] = vf;
#endif
    }
  }
#if HAVE_ASYNC_LOAD
  // Barriers do not wait on memory counters: drain our async copies first.
  asm volatile("s_wait_asynccnt 0x0" ::: "memory");
#endif
  __syncthreads();

  const v8f vzero = {0.f, 0.f, 0.f, 0.f, 0.f, 0.f, 0.f, 0.f};

  for (int blk = 0; blk < 4; ++blk) {
    const int w0 = (wv * 4 + blk) * 16;

    // ---- GEMM1: S[w0:w0+16, 0:256] = Q^T K (K-dim = H = 32, 1 WMMA/tile)
    v16bf aq = lds_ld_a(&sQt[w0 + l16][0], half);
    v8f s[16];
#pragma unroll
    for (int t = 0; t < 16; ++t) {
      v16bf bk = lds_ld_b(&sKt[t * 16 + l16][0], half);
      s[t] = wmma_bf16(aq, bk, vzero);
    }

    // ---- Row softmax. C-layout: VGPR r <-> rows (r, r+8), lane%16 <-> col.
    v8f mx = s[0];
#pragma unroll
    for (int t = 1; t < 16; ++t) {
#pragma unroll
      for (int r = 0; r < 8; ++r) mx[r] = fmaxf(mx[r], s[t][r]);
    }
#pragma unroll
    for (int off = 8; off >= 1; off >>= 1) {   // reduce within 16-lane halves
#pragma unroll
      for (int r = 0; r < 8; ++r) mx[r] = fmaxf(mx[r], __shfl_xor(mx[r], off, 32));
    }
    v8f sum = vzero;
#pragma unroll
    for (int t = 0; t < 16; ++t) {
#pragma unroll
      for (int r = 0; r < 8; ++r) {
        float e = exp2f((s[t][r] - mx[r]) * 1.44269504f);
        s[t][r] = e;
        sum[r] += e;
      }
    }
#pragma unroll
    for (int off = 8; off >= 1; off >>= 1) {
#pragma unroll
      for (int r = 0; r < 8; ++r) sum[r] += __shfl_xor(sum[r], off, 32);
    }
    v8f rn;
#pragma unroll
    for (int r = 0; r < 8; ++r) rn[r] = 1.0f / sum[r];

    // ---- P (bf16) to per-wave LDS in A-operand row-major [m][v] layout.
#pragma unroll
    for (int t = 0; t < 16; ++t) {
#pragma unroll
      for (int r = 0; r < 8; ++r)
        sP[wv][r + 8 * half][t * 16 + l16] = (__bf16)(s[t][r] * rn[r]);
    }
    // Cross-lane LDS write -> read within the wave: drain DS counter.
    asm volatile("s_wait_dscnt 0x0" ::: "memory");

    // ---- GEMM2: out^T[w0:w0+16, 0:32] = P (16x256) * V^T (256x32)
    v8f o0 = vzero, o1 = vzero;
#pragma unroll
    for (int kv = 0; kv < 8; ++kv) {
      v16bf ap = lds_ld_a(&sP[wv][l16][kv * 32], half);
      v16bf b0 = lds_ld_b(&sVb[l16][kv * 32], half);        // n-tile h=0..15
      v16bf b1 = lds_ld_b(&sVb[16 + l16][kv * 32], half);   // n-tile h=16..31
      o0 = wmma_bf16(ap, b0, o0);
      o1 = wmma_bf16(ap, b1, o1);
    }

    // ---- x = out + V staged to LDS (element (w = w0+r+8*half, h = col)).
#pragma unroll
    for (int r = 0; r < 8; ++r) {
      int w = w0 + r + 8 * half;
      sX[l16][w]      = o0[r] + sVf[l16][w];
      sX[16 + l16][w] = o1[r] + sVf[16 + l16][w];
    }
  }
  __syncthreads();

  // ---- Store x (async from LDS when available) + per-slice sum/sumsq.
  float lsum = 0.f, lsq = 0.f;
  const float4* xflat4 = (const float4*)&sX[0][0];
  float4* dst4 = (float4*)(Xout + (size_t)slice * SLICE);
  for (int i = tid; i < SLICE / 4; i += 128) {
    float4 x = xflat4[i];
#if HAVE_ASYNC_STORE
    __builtin_amdgcn_global_store_async_from_lds_b128(
        (AS1 v4i*)(dst4 + i),
        (AS3 v4i*)((char*)&sX[0][0] + (size_t)i * 16), 0, 0);
#else
    dst4[i] = x;
#endif
    lsum += (x.x + x.y) + (x.z + x.w);
    lsq  += (x.x * x.x + x.y * x.y) + (x.z * x.z + x.w * x.w);
  }
#pragma unroll
  for (int off = 16; off >= 1; off >>= 1) {
    lsum += __shfl_xor(lsum, off, 32);
    lsq  += __shfl_xor(lsq, off, 32);
  }
  if (lane == 0) { sRed[wv] = lsum; sRed[4 + wv] = lsq; }
  __syncthreads();
  if (tid == 0) {
    ws[slice]          = sRed[0] + sRed[1] + sRed[2] + sRed[3];
    ws[NSLICE + slice] = sRed[4] + sRed[5] + sRed[6] + sRed[7];
  }
#if HAVE_ASYNC_STORE
  // S_ENDPGM performs an implicit wait-idle, but drain explicitly for clarity.
  asm volatile("s_wait_asynccnt 0x0" ::: "memory");
#endif
}

__global__ __launch_bounds__(256) void bn_swish_phase2(
    float* __restrict__ X, const float* __restrict__ ws,
    const float* __restrict__ gamma, const float* __restrict__ beta) {
  const int slice = blockIdx.x;
  const int c = slice & (CH - 1);

  float s = 0.f, sq = 0.f;
#pragma unroll
  for (int b = 0; b < BATCH; ++b) {
    s  += ws[b * CH + c];
    sq += ws[NSLICE + b * CH + c];
  }
  const float invN = 1.0f / (float)(BATCH * SLICE);
  const float mean = s * invN;
  const float var  = sq * invN - mean * mean;
  const float rstd = rsqrtf(var + 1e-5f);
  const float sc = gamma[c] * rstd;
  const float sh = beta[c] - mean * sc;

  float4* p = (float4*)(X + (size_t)slice * SLICE);
  for (int i = threadIdx.x; i < SLICE / 4; i += 256) {
    float4 x = p[i];
    float xs[4] = {x.x, x.y, x.z, x.w};
#pragma unroll
    for (int j = 0; j < 4; ++j) {
      float xn = xs[j] * sc + sh;
      xs[j] = xn / (1.0f + __expf(-xn));   // swish
    }
    p[i] = make_float4(xs[0], xs[1], xs[2], xs[3]);
  }
}

extern "C" void kernel_launch(void* const* d_in, const int* in_sizes, int n_in,
                              void* d_out, int out_size, void* d_ws, size_t ws_size,
                              hipStream_t stream) {
  const float* q     = (const float*)d_in[0];
  const float* k     = (const float*)d_in[1];
  const float* v     = (const float*)d_in[2];
  const float* gamma = (const float*)d_in[3];
  const float* beta  = (const float*)d_in[4];
  float* out = (float*)d_out;
  float* ws  = (float*)d_ws;   // [0,1024): sum per slice, [1024,2048): sumsq

  attn_phase1<<<NSLICE, 128, 0, stream>>>(q, k, v, out, ws);
  bn_swish_phase2<<<NSLICE, 256, 0, stream>>>(out, ws, gamma, beta);
}
